// NashT5Block_16045997818097
// MI455X (gfx1250) — matrix operand order repaired and checked by reading, not compiled, not run on hardware
//
#include <hip/hip_runtime.h>
#include <hip/hip_bf16.h>

// ---------------------------------------------------------------------------
// T5 encoder block for MI455X (gfx1250, wave32, WMMA).
// bf16 WMMA (v_wmma_f32_16x16x32_bf16) with f32 accumulation for all GEMMs
// and flash-style attention. Global->LDS staging uses CDNA5 async copies
// (global_load_async_to_lds_b128 / s_wait_asynccnt) with double buffering.
// ---------------------------------------------------------------------------

typedef __bf16 bf16;
typedef __attribute__((ext_vector_type(16))) __bf16 bf16x16;
typedef __attribute__((ext_vector_type(8)))  float  fx8;

union FragU { uint4 u[2]; bf16x16 v; };

__device__ inline fx8 wmma_bf16(bf16x16 a, bf16x16 b, fx8 c) {
  // 8 args: (neg_a, A, neg_b, B, c_mod, C, reuse_a, reuse_b)
  return __builtin_amdgcn_wmma_f32_16x16x32_bf16(
      false, a, false, b, (short)0, c, false, false);
}

__device__ inline fx8 zero_fx8() { fx8 z = {}; return z; }

// LDS byte address for async-to-LDS ops: low 32 bits of the shared-aperture
// flat address are the LDS address (ISA 10.2 aperture mapping).
__device__ inline unsigned lds_addr(const void* p) {
  return (unsigned)(unsigned long long)p;
}

// CDNA5 async global->LDS copy of 16 bytes per lane (ASYNCcnt-tracked).
__device__ inline void async_ld16(unsigned lds, const bf16* g) {
  asm volatile("global_load_async_to_lds_b128 %0, %1, off"
               :
               : "v"(lds), "v"((unsigned long long)g)
               : "memory");
}
__device__ inline void wait_async0() {
  asm volatile("s_wait_asynccnt 0" ::: "memory");
}

#define D_MODEL 1024
#define N_HEADS 16
#define D_KV    64
#define INNER   1024
#define D_FF    4096
#define SEQ     1024
#define BATCH   2
#define ROWS    (BATCH * SEQ)   // 2048

// ---------------------------------------------------------------------------
// f32 -> bf16 conversion
// ---------------------------------------------------------------------------
__global__ void f32_to_bf16_kernel(const float* __restrict__ in,
                                   bf16* __restrict__ out, int n) {
  int i = blockIdx.x * blockDim.x + threadIdx.x;
  if (i < n) out[i] = (bf16)in[i];
}

// ---------------------------------------------------------------------------
// T5 RMSNorm (no mean subtraction, no bias), f32 in -> bf16 out.
// ---------------------------------------------------------------------------
__global__ __launch_bounds__(256)
void rmsnorm_bf16_kernel(const float* __restrict__ x,
                         const float* __restrict__ w,
                         bf16* __restrict__ out) {
  __shared__ float red[256];
  const int row = blockIdx.x;
  const float* xr = x + (size_t)row * D_MODEL;
  float v[4];
  float s = 0.f;
#pragma unroll
  for (int i = 0; i < 4; i++) {
    v[i] = xr[threadIdx.x + i * 256];
    s += v[i] * v[i];
  }
  red[threadIdx.x] = s;
  __syncthreads();
  for (int off = 128; off > 0; off >>= 1) {
    if (threadIdx.x < off) red[threadIdx.x] += red[threadIdx.x + off];
    __syncthreads();
  }
  const float rinv = rsqrtf(red[0] * (1.0f / D_MODEL) + 1e-6f);
#pragma unroll
  for (int i = 0; i < 4; i++) {
    int c = threadIdx.x + i * 256;
    out[(size_t)row * D_MODEL + c] = (bf16)(v[i] * rinv * w[c]);
  }
}

// ---------------------------------------------------------------------------
// Tiled bf16 WMMA GEMM:  C[M,N] = A[M,K] @ W[N,K]^T  (+ epilogue)
//   EPI 0: bf16 out | EPI 1: bf16 ReLU out | EPI 2: f32 out = residual + acc
// Block: 256 threads = 8 waves; tile 128x128; K-step 32; double-buffered
// async global->LDS staging; wave computes 32x64 = 2x4 WMMA tiles.
// ---------------------------------------------------------------------------
template <int EPI>
__global__ __launch_bounds__(256)
void gemm_wmma_kernel(const bf16* __restrict__ A, const bf16* __restrict__ W,
                      void* __restrict__ Out, const float* __restrict__ Res,
                      int M, int N, int K) {
  __shared__ __align__(16) bf16 As[2][128 * 40];
  __shared__ __align__(16) bf16 Bs[2][128 * 40];

  const int tid  = threadIdx.x;
  const int lane = tid & 31;
  const int wave = tid >> 5;
  const int half = lane >> 4;      // lane half selects 16-wide K-chunk (contiguous k order)
  const int ln   = lane & 15;
  const int bm = blockIdx.y * 128;
  const int bn = blockIdx.x * 128;
  const int wm = (wave >> 1) * 32; // 0,32,64,96
  const int wn = (wave & 1) * 64;  // 0,64

  // Per-thread staging assignment: two rows (r0, r0+64), one 8-elem chunk.
  const int r0 = tid >> 2;         // 0..63
  const int r1 = r0 + 64;
  const int cc = (tid & 3) << 3;   // 0,8,16,24 (elements)

  fx8 acc[2][4];
#pragma unroll
  for (int i = 0; i < 2; i++)
#pragma unroll
    for (int j = 0; j < 4; j++) acc[i][j] = zero_fx8();

  auto issue_tiles = [&](int buf, int k0) {
    async_ld16(lds_addr(&As[buf][r0 * 40 + cc]),
               &A[(size_t)(bm + r0) * K + k0 + cc]);
    async_ld16(lds_addr(&As[buf][r1 * 40 + cc]),
               &A[(size_t)(bm + r1) * K + k0 + cc]);
    async_ld16(lds_addr(&Bs[buf][r0 * 40 + cc]),
               &W[(size_t)(bn + r0) * K + k0 + cc]);
    async_ld16(lds_addr(&Bs[buf][r1 * 40 + cc]),
               &W[(size_t)(bn + r1) * K + k0 + cc]);
  };

  issue_tiles(0, 0);
  wait_async0();
  __syncthreads();

  int cur = 0;
  for (int k0 = 0; k0 < K; k0 += 32) {
    if (k0 + 32 < K) issue_tiles(cur ^ 1, k0 + 32);
    if (k0 + 64 < K) {             // warm L2 two steps ahead
      __builtin_prefetch(&A[(size_t)(bm + r0) * K + k0 + 64 + cc], 0, 1);
      __builtin_prefetch(&W[(size_t)(bn + r0) * K + k0 + 64 + cc], 0, 1);
    }

    // A fragments: lanes 0-15 K=0..15; lanes 16-31 K=16..31
    bf16x16 afr[2];
#pragma unroll
    for (int mi = 0; mi < 2; mi++) {
      const bf16* p = &As[cur][(wm + mi * 16 + ln) * 40 + (half ? 16 : 0)];
      FragU f;
      f.u[0] = *(const uint4*)p;
      f.u[1] = *(const uint4*)(p + 8);
      afr[mi] = f.v;
    }
    // B fragments: lanes 0-15 K=0..15; lanes 16-31 K=16..31
    bf16x16 bfr[4];
#pragma unroll
    for (int ni = 0; ni < 4; ni++) {
      const bf16* p = &Bs[cur][(wn + ni * 16 + ln) * 40 + (half ? 16 : 0)];
      FragU f;
      f.u[0] = *(const uint4*)p;
      f.u[1] = *(const uint4*)(p + 8);
      bfr[ni] = f.v;
    }
#pragma unroll
    for (int mi = 0; mi < 2; mi++)
#pragma unroll
      for (int ni = 0; ni < 4; ni++)
        acc[mi][ni] = wmma_bf16(afr[mi], bfr[ni], acc[mi][ni]);

    wait_async0();     // our async writes into buf^1 have landed in LDS
    __syncthreads();   // everyone done reading buf before it is rewritten
    cur ^= 1;
  }

  // Epilogue. C layout: VGPR r -> row r + 8*half, lane%16 -> col.
#pragma unroll
  for (int mi = 0; mi < 2; mi++)
#pragma unroll
    for (int ni = 0; ni < 4; ni++)
#pragma unroll
      for (int r = 0; r < 8; r++) {
        int row = bm + wm + mi * 16 + r + 8 * half;
        int col = bn + wn + ni * 16 + ln;
        size_t idx = (size_t)row * N + col;
        float v = acc[mi][ni][r];
        if (EPI == 0) {
          ((bf16*)Out)[idx] = (bf16)v;
        } else if (EPI == 1) {
          ((bf16*)Out)[idx] = (bf16)(v > 0.f ? v : 0.f);
        } else {
          ((float*)Out)[idx] = Res[idx] + v;
        }
      }
}

// ---------------------------------------------------------------------------
// T5 relative position bias bucket (bidirectional, 32 buckets, max_dist 128)
// ---------------------------------------------------------------------------
__device__ inline float t5_bias(const float* __restrict__ rb, int h, int qpos,
                                int kpos) {
  int rel = kpos - qpos;
  int bucket = (rel > 0) ? 16 : 0;
  int rp = rel < 0 ? -rel : rel;
  int add;
  if (rp < 8) {
    add = rp;
  } else {
    // 8 + log(rp/8)/log(16) * 8 ; 8/log(16) = 2.885390082
    add = 8 + (int)(__logf((float)rp * 0.125f) * 2.885390082f);
    if (add > 15) add = 15;
  }
  return rb[(bucket + add) * N_HEADS + h];
}

// ---------------------------------------------------------------------------
// Flash-style attention with relative-position bias (no 1/sqrt(d) scale).
// Grid: (S/64, B*H). Block: 128 threads = 4 waves; wave owns 16 q rows.
// KV tile = 32 positions; online softmax; K tile staged via async copies;
// V tile transposed through registers into LDS for the P@V WMMA B-fragments.
// ---------------------------------------------------------------------------
__global__ __launch_bounds__(128)
void flash_attn_kernel(const bf16* __restrict__ qb, const bf16* __restrict__ kb,
                       const bf16* __restrict__ vb,
                       const float* __restrict__ relbias,
                       bf16* __restrict__ ctx) {
  __shared__ __align__(16) bf16 Kt[32 * 72];       // K tile  [kv][d]
  __shared__ __align__(16) bf16 Vt[64 * 40];       // V^T tile [d][kv]
  __shared__ __align__(16) bf16 Ps[4 * 16 * 40];   // per-wave P staging
  __shared__ float rb[32 * N_HEADS];               // 512 floats

  const int tid  = threadIdx.x;
  const int lane = tid & 31;
  const int wave = tid >> 5;
  const int half = lane >> 4;
  const int ln   = lane & 15;
  const int bh = blockIdx.y;          // 0..31
  const int b  = bh >> 4;
  const int h  = bh & 15;
  const int q0 = blockIdx.x * 64 + wave * 16;
  const size_t rowbase = (size_t)b * SEQ;

  for (int i = tid; i < 32 * N_HEADS; i += 128) rb[i] = relbias[i];

  // Q fragments for this wave's 16 rows: K-dim = d (0..31 and 32..63).
  bf16x16 aq0, aq1;
  {
    const bf16* p = qb + (rowbase + q0 + ln) * INNER + h * D_KV;
    const int base = half ? 16 : 0;
    FragU f;
    f.u[0] = *(const uint4*)(p + base);
    f.u[1] = *(const uint4*)(p + base + 8);
    aq0 = f.v;
    f.u[0] = *(const uint4*)(p + 32 + base);
    f.u[1] = *(const uint4*)(p + 32 + base + 8);
    aq1 = f.v;
  }

  float mrow[8], lrow[8];
  fx8 o[4];
#pragma unroll
  for (int r = 0; r < 8; r++) { mrow[r] = -1e30f; lrow[r] = 0.f; }
#pragma unroll
  for (int t = 0; t < 4; t++) o[t] = zero_fx8();

  bf16* Pw = &Ps[wave * 16 * 40];

  for (int kt = 0; kt < 32; kt++) {
    const int kv0 = kt * 32;
    // Stage K tile [32][64] via async copies; V^T tile via register transpose.
    {
      const int r = tid >> 2;          // 0..31 kv row
      const int c = (tid & 3) * 16;    // 0,16,32,48 d offset
      const bf16* pk = kb + (rowbase + kv0 + r) * INNER + h * D_KV + c;
      async_ld16(lds_addr(&Kt[r * 72 + c]), pk);
      async_ld16(lds_addr(&Kt[r * 72 + c + 8]), pk + 8);
      const bf16* pv = vb + (rowbase + kv0 + r) * INNER + h * D_KV + c;
      bf16 tmp[16];
      *(uint4*)&tmp[0] = *(const uint4*)pv;
      *(uint4*)&tmp[8] = *(const uint4*)(pv + 8);
#pragma unroll
      for (int e = 0; e < 16; e++) Vt[(c + e) * 40 + r] = tmp[e];
    }
    wait_async0();
    __syncthreads();

    // Scores: two 16-column subtiles of the 16x32 score block.
    fx8 c0 = zero_fx8(), c1 = zero_fx8();
    {
      const bf16* p0 = &Kt[(0 + ln) * 72 + (half ? 16 : 0)];
      FragU f;
      f.u[0] = *(const uint4*)p0;        f.u[1] = *(const uint4*)(p0 + 8);
      bf16x16 bk = f.v;
      c0 = wmma_bf16(aq0, bk, c0);
      f.u[0] = *(const uint4*)(p0 + 32); f.u[1] = *(const uint4*)(p0 + 40);
      bk = f.v;
      c0 = wmma_bf16(aq1, bk, c0);

      const bf16* p1 = &Kt[(16 + ln) * 72 + (half ? 16 : 0)];
      f.u[0] = *(const uint4*)p1;        f.u[1] = *(const uint4*)(p1 + 8);
      bk = f.v;
      c1 = wmma_bf16(aq0, bk, c1);
      f.u[0] = *(const uint4*)(p1 + 32); f.u[1] = *(const uint4*)(p1 + 40);
      bk = f.v;
      c1 = wmma_bf16(aq1, bk, c1);
    }

    // Bias + online softmax update (rows live across 16 lanes of each half).
    fx8 sv;
#pragma unroll
    for (int r = 0; r < 8; r++) {
      const int m = r + 8 * half;
      const int qpos = q0 + m;
      c0[r] += t5_bias(rb, h, qpos, kv0 + ln);
      c1[r] += t5_bias(rb, h, qpos, kv0 + 16 + ln);

      float rm = fmaxf(c0[r], c1[r]);
#pragma unroll
      for (int off = 1; off < 16; off <<= 1)
        rm = fmaxf(rm, __shfl_xor(rm, off, 16));
      const float mn = fmaxf(mrow[r], rm);
      const float sc = __expf(mrow[r] - mn);
      c0[r] = __expf(c0[r] - mn);
      c1[r] = __expf(c1[r] - mn);
      float rs = c0[r] + c1[r];
#pragma unroll
      for (int off = 1; off < 16; off <<= 1)
        rs += __shfl_xor(rs, off, 16);
      lrow[r] = lrow[r] * sc + rs;
      mrow[r] = mn;
      sv[r] = sc;
    }
#pragma unroll
    for (int t = 0; t < 4; t++) o[t] *= sv;

    // Restage P (bf16) into wave-private LDS row-major [16][32].
#pragma unroll
    for (int r = 0; r < 8; r++) {
      const int m = r + 8 * half;
      Pw[m * 40 + ln]      = (bf16)c0[r];
      Pw[m * 40 + 16 + ln] = (bf16)c1[r];
    }

    // P A-fragment (rows = q rows, K = 32 kv positions).
    bf16x16 ap;
    {
      const bf16* p = &Pw[ln * 40 + (half ? 16 : 0)];
      FragU f;
      f.u[0] = *(const uint4*)p;
      f.u[1] = *(const uint4*)(p + 8);
      ap = f.v;
    }
    // O += P @ V ; B fragments from transposed V tile.
#pragma unroll
    for (int t = 0; t < 4; t++) {
      const bf16* p = &Vt[(t * 16 + ln) * 40 + (half ? 16 : 0)];
      FragU f;
      f.u[0] = *(const uint4*)p;
      f.u[1] = *(const uint4*)(p + 8);
      o[t] = wmma_bf16(ap, f.v, o[t]);
    }
    __syncthreads();
  }

  // Normalize and store ctx (bf16) in [BS, INNER] layout.
#pragma unroll
  for (int t = 0; t < 4; t++)
#pragma unroll
    for (int r = 0; r < 8; r++) {
      const int m = r + 8 * half;
      ctx[(rowbase + q0 + m) * INNER + h * D_KV + t * 16 + ln] =
          (bf16)(o[t][r] / lrow[r]);
    }
}

// ---------------------------------------------------------------------------
// Launcher
// ---------------------------------------------------------------------------
extern "C" void kernel_launch(void* const* d_in, const int* in_sizes, int n_in,
                              void* d_out, int out_size, void* d_ws,
                              size_t ws_size, hipStream_t stream) {
  const float* hidden  = (const float*)d_in[0];
  const float* ln1_w   = (const float*)d_in[1];
  const float* wq      = (const float*)d_in[2];
  const float* wk      = (const float*)d_in[3];
  const float* wv      = (const float*)d_in[4];
  const float* wo      = (const float*)d_in[5];
  const float* relbias = (const float*)d_in[6];
  const float* ln2_w   = (const float*)d_in[7];
  const float* wi      = (const float*)d_in[8];
  const float* wo_ff   = (const float*)d_in[9];

  char* ws = (char*)d_ws;
  bf16* xb   = (bf16*)ws;  ws += (size_t)ROWS * D_MODEL * 2;
  bf16* qb   = (bf16*)ws;  ws += (size_t)ROWS * INNER * 2;
  bf16* kbuf = (bf16*)ws;  ws += (size_t)ROWS * INNER * 2;
  bf16* vbuf = (bf16*)ws;  ws += (size_t)ROWS * INNER * 2;
  bf16* ctxb = (bf16*)ws;  ws += (size_t)ROWS * INNER * 2;
  float* hbuf = (float*)ws; ws += (size_t)ROWS * D_MODEL * 4;
  bf16* yb   = (bf16*)ws;  ws += (size_t)ROWS * D_MODEL * 2;
  bf16* ffb  = (bf16*)ws;  ws += (size_t)ROWS * D_FF * 2;
  bf16* wqb  = (bf16*)ws;  ws += (size_t)INNER * D_MODEL * 2;
  bf16* wkb  = (bf16*)ws;  ws += (size_t)INNER * D_MODEL * 2;
  bf16* wvb  = (bf16*)ws;  ws += (size_t)INNER * D_MODEL * 2;
  bf16* wob  = (bf16*)ws;  ws += (size_t)D_MODEL * INNER * 2;
  bf16* wib  = (bf16*)ws;  ws += (size_t)D_FF * D_MODEL * 2;
  bf16* wofb = (bf16*)ws;  ws += (size_t)D_MODEL * D_FF * 2;

  auto cvt = [&](const float* src, bf16* dst, int n) {
    f32_to_bf16_kernel<<<(n + 255) / 256, 256, 0, stream>>>(src, dst, n);
  };
  cvt(wq, wqb, INNER * D_MODEL);
  cvt(wk, wkb, INNER * D_MODEL);
  cvt(wv, wvb, INNER * D_MODEL);
  cvt(wo, wob, D_MODEL * INNER);
  cvt(wi, wib, D_FF * D_MODEL);
  cvt(wo_ff, wofb, D_MODEL * D_FF);

  // x = RMSNorm(hidden, ln1)
  rmsnorm_bf16_kernel<<<ROWS, 256, 0, stream>>>(hidden, ln1_w, xb);

  // Q, K, V projections
  dim3 gqkv(INNER / 128, ROWS / 128);
  gemm_wmma_kernel<0><<<gqkv, 256, 0, stream>>>(xb, wqb, qb, nullptr,
                                                ROWS, INNER, D_MODEL);
  gemm_wmma_kernel<0><<<gqkv, 256, 0, stream>>>(xb, wkb, kbuf, nullptr,
                                                ROWS, INNER, D_MODEL);
  gemm_wmma_kernel<0><<<gqkv, 256, 0, stream>>>(xb, wvb, vbuf, nullptr,
                                                ROWS, INNER, D_MODEL);

  // Attention with relative position bias
  dim3 gattn(SEQ / 64, BATCH * N_HEADS);
  flash_attn_kernel<<<gattn, 128, 0, stream>>>(qb, kbuf, vbuf, relbias, ctxb);

  // h = hidden + ctx @ wo^T
  dim3 go(D_MODEL / 128, ROWS / 128);
  gemm_wmma_kernel<2><<<go, 256, 0, stream>>>(ctxb, wob, hbuf, hidden,
                                              ROWS, D_MODEL, INNER);

  // y = RMSNorm(h, ln2)
  rmsnorm_bf16_kernel<<<ROWS, 256, 0, stream>>>(hbuf, ln2_w, yb);

  // ff = relu(y @ wi^T)
  dim3 gff(D_FF / 128, ROWS / 128);
  gemm_wmma_kernel<1><<<gff, 256, 0, stream>>>(yb, wib, ffb, nullptr,
                                               ROWS, D_FF, D_MODEL);

  // out = h + ff @ wo_ff^T
  dim3 gout(D_MODEL / 128, ROWS / 128);
  gemm_wmma_kernel<2><<<gout, 256, 0, stream>>>(ffb, wofb, (float*)d_out, hbuf,
                                                ROWS, D_MODEL, D_FF);
}
